// LocalEncoder_33603824124645
// MI455X (gfx1250) — compile-verified, matrix operand order checked by reading, not run on hardware
//
#include <hip/hip_runtime.h>

typedef unsigned short u16;
typedef __attribute__((ext_vector_type(16))) __bf16 v16bf;
typedef __attribute__((ext_vector_type(8)))  float  v8f;

// ---------------------------------------------------------------- helpers ---

__device__ __forceinline__ u16 f2bf(float x) {
  unsigned u = __builtin_bit_cast(unsigned, x);
  unsigned r = (u + 0x7FFFu + ((u >> 16) & 1u)) >> 16;
  return (u16)r;
}
__device__ __forceinline__ float bf2f(u16 h) {
  unsigned u = ((unsigned)h) << 16;
  return __builtin_bit_cast(float, u);
}

// Fragment where element [r][k] = p[r*ld + k]  (A row-major, or B given as [N][K])
// CDNA5 16-bit A layout: lane&15 = row, lanes<16 hold K 0-7/16-23, lanes>=16 hold 8-15/24-31.
__device__ __forceinline__ v16bf frag_ld_nk(const u16* __restrict__ p, int ld) {
  int lane = threadIdx.x & 31;
  const u16* q = p + (lane & 15) * ld + ((lane >> 4) << 3);
  v16bf f;
#pragma unroll
  for (int j = 0; j < 8; ++j) {
    f[j]     = __builtin_bit_cast(__bf16, q[j]);
    f[j + 8] = __builtin_bit_cast(__bf16, q[16 + j]);
  }
  return f;
}

// Fragment where element [k][n] = p[k*ld + n]  (B given as [K][N])
__device__ __forceinline__ v16bf frag_ld_kn(const u16* __restrict__ p, int ld) {
  int lane = threadIdx.x & 31;
  const u16* q = p + (((lane >> 4) << 3)) * ld + (lane & 15);
  v16bf f;
#pragma unroll
  for (int j = 0; j < 8; ++j) {
    f[j]     = __builtin_bit_cast(__bf16, q[j * ld]);
    f[j + 8] = __builtin_bit_cast(__bf16, q[(16 + j) * ld]);
  }
  return f;
}

__device__ __forceinline__ v8f wmma_bf16(v16bf a, v16bf b, v8f c) {
  return __builtin_amdgcn_wmma_f32_16x16x32_bf16(false, a, false, b, (short)0, c,
                                                 false, false);
}

// ---------------------------------------------------------------- embed -----
// emb[p,l,:] = (l < len) ? byte_emb[tok] + local_pos[l] : 0   -> bf16
__global__ __launch_bounds__(256) void embed_k(const int* __restrict__ tokens,
                                               const int* __restrict__ lengths,
                                               const float* __restrict__ bemb,
                                               const float* __restrict__ lpos,
                                               u16* __restrict__ out, int p0) {
  int r = blockIdx.x;            // local row, 0 .. 256*32-1
  int p = p0 + (r >> 5), l = r & 31;
  int len = lengths[p];
  int tok = tokens[p * 32 + l];
  bool valid = (l < len);
  for (int c = threadIdx.x; c < 512; c += 256) {
    float v = valid ? (bemb[tok * 512 + c] + lpos[l * 512 + c]) : 0.0f;
    out[(size_t)r * 512 + c] = f2bf(v);
  }
}

// ---------------------------------------------------------------- gemm ------
// C[M,N] = A(bf16)[M,K] @ W(f32)[N,K]^T + bias   with epilogue variants.
// grid = (N/256, M/128), block = 256 (8 waves as 2x4; each wave owns 64x64).
// K staged 64 at a time; per stage each wave runs 32 WMMAs with 4x fragment reuse.
// EPI: 0 = bf16 out, 1 = bf16 gelu out, 2 = f32 out +=, 3 = f32 out = v + add
template <int EPI>
__global__ __launch_bounds__(256) void gemm_bf16_k(
    const u16* __restrict__ A, int lda, const float* __restrict__ W, int ldw,
    const float* __restrict__ bias, void* __restrict__ out, int ldo,
    const float* __restrict__ add, int ldadd, int K) {
  __shared__ __align__(16) u16 As[128 * 72];
  __shared__ __align__(16) u16 Ws[256 * 72];
  int tid = threadIdx.x, lane = tid & 31, wid = tid >> 5;
  int wm = wid >> 2, wn = wid & 3;          // 2 x 4 wave grid
  int Mb = blockIdx.y * 128, Nb = blockIdx.x * 256;
  v8f acc[4][4] = {};
  // A staging: 2 threads per row, 32 bf16 (4x uint4) each
  int ar = tid >> 1, ac = (tid & 1) * 32;
  const u16* aptr = A + (size_t)(Mb + ar) * lda + ac;
  u16* adst = &As[ar * 72 + ac];
  // W staging: 1 thread per row, 64 f32 -> bf16
  const float* wptr = W + (size_t)(Nb + tid) * ldw;
  u16* wdst = &Ws[tid * 72];
  for (int k0 = 0; k0 < K; k0 += 64) {
    {
      const uint4* s = (const uint4*)(aptr + k0);
      uint4* d = (uint4*)adst;
#pragma unroll
      for (int j = 0; j < 4; ++j) d[j] = s[j];
    }
    if (k0 + 64 < K) __builtin_prefetch(wptr + k0 + 64, 0, 1);
#pragma unroll
    for (int j = 0; j < 64; ++j) wdst[j] = f2bf(wptr[k0 + j]);
    __syncthreads();
#pragma unroll
    for (int kk = 0; kk < 64; kk += 32) {
      v16bf bf[4];
#pragma unroll
      for (int nt = 0; nt < 4; ++nt)
        bf[nt] = frag_ld_nk(&Ws[(wn * 64 + nt * 16) * 72 + kk], 72);
#pragma unroll
      for (int mt = 0; mt < 4; ++mt) {
        v16bf a = frag_ld_nk(&As[(wm * 64 + mt * 16) * 72 + kk], 72);
#pragma unroll
        for (int nt = 0; nt < 4; ++nt) acc[mt][nt] = wmma_bf16(a, bf[nt], acc[mt][nt]);
      }
    }
    __syncthreads();
  }
  int n0 = lane & 15, mo = (lane >> 4) * 8;
#pragma unroll
  for (int nt = 0; nt < 4; ++nt) {
    int gn = Nb + wn * 64 + nt * 16 + n0;
    float bv = bias[gn];
#pragma unroll
    for (int mt = 0; mt < 4; ++mt) {
#pragma unroll
      for (int rr = 0; rr < 8; ++rr) {
        int gm = Mb + wm * 64 + mt * 16 + mo + rr;
        float vv = acc[mt][nt][rr] + bv;
        if (EPI == 0) {
          ((u16*)out)[(size_t)gm * ldo + gn] = f2bf(vv);
        } else if (EPI == 1) {
          float g = 0.5f * vv * (1.0f + erff(vv * 0.70710678118654752f));
          ((u16*)out)[(size_t)gm * ldo + gn] = f2bf(g);
        } else if (EPI == 2) {
          ((float*)out)[(size_t)gm * ldo + gn] += vv;
        } else {
          ((float*)out)[(size_t)gm * ldo + gn] = vv + add[(size_t)gm * ldadd + gn];
        }
      }
    }
  }
}

// ---------------------------------------------------------- patch attention -
// One block per patch. qkvp rows: [q 0..511 | k 512..1023 | v 1024..1535],
// 4 heads of d=128.  Produces masked-mean context vector [512] (bf16).
__global__ __launch_bounds__(256) void patch_attn(const u16* __restrict__ qkvp,
                                                  const int* __restrict__ lengths,
                                                  int p0, u16* __restrict__ cmean) {
  int pl = blockIdx.x;
  int p = p0 + pl;
  int len = lengths[p];
  const u16* base = qkvp + (size_t)pl * 32 * 1536;
  __shared__ float Sh[4][32][32];
  __shared__ __align__(16) u16 Ph[4][32][32];
  __shared__ __align__(16) u16 Ctx[32][512];
  int tid = threadIdx.x, lane = tid & 31, wid = tid >> 5;
  int n0 = lane & 15, mo = (lane >> 4) * 8;
  // S = q @ k^T : 4 heads x 2x2 tiles = 16 tasks over 8 waves, K=128
  for (int t = wid; t < 16; t += 8) {
    int hh = t >> 2, mt = (t >> 1) & 1, nt = t & 1;
    v8f acc = {};
#pragma unroll
    for (int ks = 0; ks < 128; ks += 32) {
      v16bf a = frag_ld_nk(base + (size_t)(mt * 16) * 1536 + hh * 128 + ks, 1536);
      v16bf b = frag_ld_nk(base + (size_t)(nt * 16) * 1536 + 512 + hh * 128 + ks, 1536);
      acc = wmma_bf16(a, b, acc);
    }
#pragma unroll
    for (int rr = 0; rr < 8; ++rr) Sh[hh][mt * 16 + mo + rr][nt * 16 + n0] = acc[rr];
  }
  __syncthreads();
  // masked softmax over keys, one row per thread (4 heads x 32 rows)
  if (tid < 128) {
    int hh = tid >> 5, m = tid & 31;
    const float scale = 0.08838834764831845f;  // 1/sqrt(128)
    float sv[32];
    float mx = -1e30f;
#pragma unroll
    for (int j = 0; j < 32; ++j) {
      float s = (j < len) ? Sh[hh][m][j] * scale : -1e9f;
      sv[j] = s;
      mx = fmaxf(mx, s);
    }
    float sum = 0.0f;
#pragma unroll
    for (int j = 0; j < 32; ++j) { sv[j] = __expf(sv[j] - mx); sum += sv[j]; }
    float inv = 1.0f / sum;
#pragma unroll
    for (int j = 0; j < 32; ++j) Ph[hh][m][j] = f2bf(sv[j] * inv);
  }
  __syncthreads();
  // ctx = P @ v : 4 heads x 2x8 tiles = 64 tasks, K=32
  for (int t = wid; t < 64; t += 8) {
    int hh = t >> 4, mt = (t >> 3) & 1, nt = t & 7;
    v16bf a = frag_ld_nk(&Ph[hh][mt * 16][0], 32);
    v16bf b = frag_ld_kn(base + 1024 + hh * 128 + nt * 16, 1536);
    v8f z = {};
    v8f acc = wmma_bf16(a, b, z);
#pragma unroll
    for (int rr = 0; rr < 8; ++rr)
      Ctx[mt * 16 + mo + rr][hh * 128 + nt * 16 + n0] = f2bf(acc[rr]);
  }
  __syncthreads();
  float flen = (float)len;
  for (int c = tid; c < 512; c += 256) {
    float s = 0.0f;
    for (int m = 0; m < len; ++m) s += bf2f(Ctx[m][c]);
    cmean[(size_t)p * 512 + c] = f2bf(s / flen);
  }
}

// ------------------------------------------------- transformer attention ----
// S[h, Mb.., Nb..] = scale * Q @ K^T  (d=96, K-step 3x32)
__global__ __launch_bounds__(256) void attn_scores(const u16* __restrict__ qkv,
                                                   float* __restrict__ S,
                                                   float scale) {
  int h = blockIdx.z;
  int Mb = blockIdx.y * 128, Nb = blockIdx.x * 128;
  __shared__ __align__(16) u16 Qs[128 * 104];
  __shared__ __align__(16) u16 Ks[128 * 104];
  int tid = threadIdx.x, lane = tid & 31, wid = tid >> 5;
  {
    int r = tid >> 1, half = tid & 1;
    const u16* qs = qkv + (size_t)(Mb + r) * 2304 + h * 96 + half * 48;
    const u16* ks = qkv + (size_t)(Nb + r) * 2304 + 768 + h * 96 + half * 48;
    uint4* qd = (uint4*)&Qs[r * 104 + half * 48];
    uint4* kd = (uint4*)&Ks[r * 104 + half * 48];
#pragma unroll
    for (int j = 0; j < 6; ++j) {
      qd[j] = ((const uint4*)qs)[j];
      kd[j] = ((const uint4*)ks)[j];
    }
  }
  __syncthreads();
  v8f acc[8] = {};
#pragma unroll
  for (int ks3 = 0; ks3 < 3; ++ks3) {
    v16bf a = frag_ld_nk(&Qs[(wid * 16) * 104 + ks3 * 32], 104);
#pragma unroll
    for (int nt = 0; nt < 8; ++nt) {
      v16bf b = frag_ld_nk(&Ks[(nt * 16) * 104 + ks3 * 32], 104);
      acc[nt] = wmma_bf16(a, b, acc[nt]);
    }
  }
  int n0 = lane & 15, mo = (lane >> 4) * 8;
#pragma unroll
  for (int nt = 0; nt < 8; ++nt)
#pragma unroll
    for (int rr = 0; rr < 8; ++rr)
      S[(size_t)(h * 1024 + Mb + wid * 16 + mo + rr) * 1024 + Nb + nt * 16 + n0] =
          acc[nt][rr] * scale;
}

// softmax over 1024-wide f32 row, write bf16 probs in place (first half of row)
__global__ __launch_bounds__(256) void softmax_rows(float* __restrict__ S) {
  int row = blockIdx.x, tid = threadIdx.x;
  float* xr = S + (size_t)row * 1024;
  float v[4];
#pragma unroll
  for (int i = 0; i < 4; ++i) v[i] = xr[tid + i * 256];
  __shared__ float red[256];
  red[tid] = fmaxf(fmaxf(v[0], v[1]), fmaxf(v[2], v[3]));
  __syncthreads();
  for (int s = 128; s > 0; s >>= 1) {
    if (tid < s) red[tid] = fmaxf(red[tid], red[tid + s]);
    __syncthreads();
  }
  float mx = red[0];
  __syncthreads();
  float sum = 0.0f;
#pragma unroll
  for (int i = 0; i < 4; ++i) { v[i] = __expf(v[i] - mx); sum += v[i]; }
  red[tid] = sum;
  __syncthreads();
  for (int s = 128; s > 0; s >>= 1) {
    if (tid < s) red[tid] += red[tid + s];
    __syncthreads();
  }
  float inv = 1.0f / red[0];
  u16* pr = (u16*)xr;
#pragma unroll
  for (int i = 0; i < 4; ++i) pr[tid + i * 256] = f2bf(v[i] * inv);
}

// ctx[Mb.., h*96..] = P @ V   (K = 1024 streamed in 32-chunks)
__global__ __launch_bounds__(256) void attn_ctx(const float* __restrict__ S,
                                                const u16* __restrict__ qkv,
                                                u16* __restrict__ ctx) {
  int Mb = blockIdx.x * 128, h = blockIdx.y;
  __shared__ __align__(16) u16 Ps[128 * 40];
  __shared__ __align__(16) u16 Vs[32 * 104];
  int tid = threadIdx.x, lane = tid & 31, wid = tid >> 5;
  v8f acc[6] = {};
  for (int k0 = 0; k0 < 1024; k0 += 32) {
    __syncthreads();
    {
      int r = tid >> 1, half = tid & 1;
      const u16* pr = (const u16*)(S + (size_t)(h * 1024 + Mb + r) * 1024);
      uint4* d = (uint4*)&Ps[r * 40 + half * 16];
      const uint4* s = (const uint4*)(pr + k0 + half * 16);
      d[0] = s[0];
      d[1] = s[1];
    }
    for (int idx = tid; idx < 32 * 96; idx += 256) {
      int k = idx / 96, n = idx - k * 96;
      Vs[k * 104 + n] = qkv[(size_t)(k0 + k) * 2304 + 1536 + h * 96 + n];
    }
    __syncthreads();
    v16bf a = frag_ld_nk(&Ps[(wid * 16) * 40], 40);
#pragma unroll
    for (int nt = 0; nt < 6; ++nt) {
      v16bf b = frag_ld_kn(&Vs[nt * 16], 104);
      acc[nt] = wmma_bf16(a, b, acc[nt]);
    }
  }
  int n0 = lane & 15, mo = (lane >> 4) * 8;
#pragma unroll
  for (int nt = 0; nt < 6; ++nt)
#pragma unroll
    for (int rr = 0; rr < 8; ++rr)
      ctx[(size_t)(Mb + wid * 16 + mo + rr) * 768 + h * 96 + nt * 16 + n0] =
          f2bf(acc[nt][rr]);
}

// ------------------------------------------------------------ layernorm -----
template <bool F32OUT>
__global__ __launch_bounds__(256) void ln_rows(const float* __restrict__ x,
                                               const float* __restrict__ w,
                                               const float* __restrict__ b,
                                               void* __restrict__ out, int D) {
  int row = blockIdx.x, tid = threadIdx.x;
  const float* xr = x + (size_t)row * D;
  float v[3];
#pragma unroll
  for (int i = 0; i < 3; ++i) v[i] = xr[tid + i * 256];
  __shared__ float red[256];
  red[tid] = v[0] + v[1] + v[2];
  __syncthreads();
  for (int s = 128; s > 0; s >>= 1) {
    if (tid < s) red[tid] += red[tid + s];
    __syncthreads();
  }
  float mu = red[0] / (float)D;
  __syncthreads();
  float sq = 0.0f;
#pragma unroll
  for (int i = 0; i < 3; ++i) { float d = v[i] - mu; sq += d * d; }
  red[tid] = sq;
  __syncthreads();
  for (int s = 128; s > 0; s >>= 1) {
    if (tid < s) red[tid] += red[tid + s];
    __syncthreads();
  }
  float rstd = rsqrtf(red[0] / (float)D + 1e-5f);
#pragma unroll
  for (int i = 0; i < 3; ++i) {
    int c = tid + i * 256;
    float y = (v[i] - mu) * rstd * w[c] + b[c];
    if (F32OUT) ((float*)out)[(size_t)row * D + c] = y;
    else ((u16*)out)[(size_t)row * D + c] = f2bf(y);
  }
}

// ---------------------------------------------------------------- launch ----
extern "C" void kernel_launch(void* const* d_in, const int* in_sizes, int n_in,
                              void* d_out, int out_size, void* d_ws, size_t ws_size,
                              hipStream_t stream) {
  (void)in_sizes; (void)n_in; (void)out_size; (void)ws_size;
  const int* tokens = (const int*)d_in[0];
  const int* lengths = (const int*)d_in[1];
  const float* byte_emb = (const float*)d_in[2];
  const float* local_pos = (const float*)d_in[3];
  const float* patch_pos = (const float*)d_in[4];
  const float* pa_qkv_w = (const float*)d_in[5];
  const float* pa_qkv_b = (const float*)d_in[6];
  const float* pa_out_w = (const float*)d_in[7];
  const float* pa_out_b = (const float*)d_in[8];
  const float* proj_w = (const float*)d_in[9];
  const float* proj_b = (const float*)d_in[10];
  const float* Lqkv_w = (const float*)d_in[11];
  const float* Lqkv_b = (const float*)d_in[12];
  const float* Lout_w = (const float*)d_in[13];
  const float* Lout_b = (const float*)d_in[14];
  const float* Lln1_w = (const float*)d_in[15];
  const float* Lln1_b = (const float*)d_in[16];
  const float* Lln2_w = (const float*)d_in[17];
  const float* Lln2_b = (const float*)d_in[18];
  const float* Lff1_w = (const float*)d_in[19];
  const float* Lff1_b = (const float*)d_in[20];
  const float* Lff2_w = (const float*)d_in[21];
  const float* Lff2_b = (const float*)d_in[22];
  const float* lnf_w = (const float*)d_in[23];
  const float* lnf_b = (const float*)d_in[24];

  char* ws = (char*)d_ws;
  size_t off = 0;
  auto take = [&](size_t bytes) -> char* {
    char* r = ws + off;
    off = (off + bytes + 255) & ~(size_t)255;
    return r;
  };
  float* h = (float*)take((size_t)1024 * 768 * 4);
  u16* hn = (u16*)take((size_t)1024 * 768 * 2);
  u16* qkv_t = (u16*)take((size_t)1024 * 2304 * 2);
  float* S = (float*)take((size_t)8 * 1024 * 1024 * 4);   // scores / probs
  u16* actx = (u16*)take((size_t)1024 * 768 * 2);
  u16* ffb = (u16*)take((size_t)1024 * 3072 * 2);
  u16* cmean = (u16*)take((size_t)1024 * 512 * 2);
  u16* repr = (u16*)take((size_t)1024 * 512 * 2);
  // patch-stage chunk buffers alias the (not-yet-used) score buffer: 8MB+24MB = 32MB
  u16* embc = (u16*)S;
  u16* qkvp = (u16*)((char*)S + (size_t)256 * 32 * 512 * 2);

  // ---- patch stage: 4 chunks of 256 patches --------------------------------
  for (int c = 0; c < 4; ++c) {
    int p0 = c * 256;
    embed_k<<<256 * 32, 256, 0, stream>>>(tokens, lengths, byte_emb, local_pos,
                                          embc, p0);
    gemm_bf16_k<0><<<dim3(1536 / 256, 8192 / 128), 256, 0, stream>>>(
        embc, 512, pa_qkv_w, 512, pa_qkv_b, qkvp, 1536, nullptr, 0, 512);
    patch_attn<<<256, 256, 0, stream>>>(qkvp, lengths, p0, cmean);
  }
  // repr = ctx_mean @ Wo^T + bo  (linearity lets mean commute with out-proj)
  gemm_bf16_k<0><<<dim3(2, 8), 256, 0, stream>>>(cmean, 512, pa_out_w, 512,
                                                 pa_out_b, repr, 512, nullptr, 0, 512);
  // h = repr @ proj^T + proj_b + patch_pos
  gemm_bf16_k<3><<<dim3(3, 8), 256, 0, stream>>>(repr, 512, proj_w, 512, proj_b,
                                                 h, 768, patch_pos, 768, 512);

  // ---- transformer encoder -------------------------------------------------
  const float scale_tr = 0.10206207261596575f;  // 1/sqrt(96)
  for (int i = 0; i < 6; ++i) {
    ln_rows<false><<<1024, 256, 0, stream>>>(h, Lln1_w + i * 768, Lln1_b + i * 768,
                                             hn, 768);
    gemm_bf16_k<0><<<dim3(9, 8), 256, 0, stream>>>(
        hn, 768, Lqkv_w + (size_t)i * 2304 * 768, 768, Lqkv_b + i * 2304, qkv_t,
        2304, nullptr, 0, 768);
    attn_scores<<<dim3(8, 8, 8), 256, 0, stream>>>(qkv_t, S, scale_tr);
    softmax_rows<<<8192, 256, 0, stream>>>(S);
    attn_ctx<<<dim3(8, 8), 256, 0, stream>>>(S, qkv_t, actx);
    gemm_bf16_k<2><<<dim3(3, 8), 256, 0, stream>>>(
        actx, 768, Lout_w + (size_t)i * 768 * 768, 768, Lout_b + i * 768, h, 768,
        nullptr, 0, 768);
    ln_rows<false><<<1024, 256, 0, stream>>>(h, Lln2_w + i * 768, Lln2_b + i * 768,
                                             hn, 768);
    gemm_bf16_k<1><<<dim3(12, 8), 256, 0, stream>>>(
        hn, 768, Lff1_w + (size_t)i * 3072 * 768, 768, Lff1_b + i * 3072, ffb,
        3072, nullptr, 0, 768);
    gemm_bf16_k<2><<<dim3(3, 8), 256, 0, stream>>>(
        ffb, 3072, Lff2_w + (size_t)i * 768 * 3072, 3072, Lff2_b + i * 768, h,
        768, nullptr, 0, 3072);
  }
  ln_rows<true><<<1024, 256, 0, stream>>>(h, lnf_w, lnf_b, d_out, 768);
}
